// FnnMlpMlpIni_13795434955130
// MI455X (gfx1250) — matrix operand order repaired and checked by reading, not compiled, not run on hardware
//
#include <hip/hip_runtime.h>
#include <hip/hip_bf16.h>

// ---------------------------------------------------------------------------
// FNN (fuzzy neural net): R=32 rules, B=16384 batch, F=128 feat, C=10 cls
// Strategy: fuse everything per (rule, 64-row batch tile); all GEMMs via
// v_wmma_f32_16x16x32_f16 with f16 activations in LDS and pre-transposed f16
// weights streamed from the 192MB L2. ~139 GFLOP total -> matrix-pipe bound.
// ---------------------------------------------------------------------------

typedef _Float16 v16h __attribute__((ext_vector_type(16)));
typedef _Float16 v8h  __attribute__((ext_vector_type(8)));
typedef float    v8f  __attribute__((ext_vector_type(8)));

constexpr int Rn   = 32;     // rules
constexpr int Bsz  = 16384;  // batch
constexpr int Fd   = 128;    // features
constexpr int Hd   = 256;    // hidden (2F)
constexpr int Ccls = 10;     // classes
constexpr int CP   = 16;     // classes padded to WMMA N
constexpr int BT   = 64;     // batch rows per block
constexpr int LDA  = 136;    // LDS stride for 128-wide buffers (+8 pad)
constexpr int LDH  = 264;    // LDS stride for 256-wide buffers (+8 pad)

// ---------------------------------------------------------------------------
// Weight transpose + fp16 conversion: WT[n][k] = W[k][n]. One elem per thread.
// ---------------------------------------------------------------------------
__global__ __launch_bounds__(256) void prep_weights(
    const float* __restrict__ W1, const float* __restrict__ W2,
    const float* __restrict__ Wc1, const float* __restrict__ Wc2,
    const float* __restrict__ Wc3,
    _Float16* __restrict__ W1T, _Float16* __restrict__ W2T,
    _Float16* __restrict__ Wc1T, _Float16* __restrict__ Wc2T,
    _Float16* __restrict__ Wc3T)
{
  long i = (long)blockIdx.x * blockDim.x + threadIdx.x;
  const long n1 = (long)Hd * Fd;        // W1T [H][F]
  const long n2 = (long)Fd * Hd;        // W2T [F][H]
  const long n3 = (long)Rn * Hd * Fd;   // Wc1T [R][H][F]
  const long n4 = (long)Rn * Fd * Hd;   // Wc2T [R][F][H]
  const long n5 = (long)Rn * CP * Fd;   // Wc3T [R][CP][F]  (cols >= C zero)
  if (i < n1) { int n = (int)(i / Fd), k = (int)(i % Fd);
    W1T[i] = (_Float16)W1[(long)k * Hd + n]; return; }
  i -= n1;
  if (i < n2) { int f = (int)(i / Hd), h = (int)(i % Hd);
    W2T[i] = (_Float16)W2[(long)h * Fd + f]; return; }
  i -= n2;
  if (i < n3) { int r = (int)(i / ((long)Hd * Fd)); long j = i % ((long)Hd * Fd);
    int n = (int)(j / Fd), k = (int)(j % Fd);
    Wc1T[i] = (_Float16)Wc1[(long)r * Fd * Hd + (long)k * Hd + n]; return; }
  i -= n3;
  if (i < n4) { int r = (int)(i / ((long)Fd * Hd)); long j = i % ((long)Fd * Hd);
    int f = (int)(j / Hd), h = (int)(j % Hd);
    Wc2T[i] = (_Float16)Wc2[(long)r * Hd * Fd + (long)h * Fd + f]; return; }
  i -= n4;
  if (i < n5) { int r = (int)(i / (CP * Fd)); int j = (int)(i % (CP * Fd));
    int c = j / Fd, f = j % Fd;
    Wc3T[i] = (_Float16)((c < Ccls) ? Wc3[(long)r * Fd * Ccls + (long)f * Ccls + c]
                                    : 0.0f);
    return; }
}

// ---------------------------------------------------------------------------
// One fused GEMM stage: sOut[BT][NOUT] = act(sIn[BT][KD] @ WT^T + bias)
// WT is [NOUT][KD] f16 in global (L2-resident). Each of 8 waves owns 16x16
// output tiles. ACT: 0=none, 1=ELU, 2=ReLU.
// A-frag (ISA 16-bit A 16x32 layout): e<8 -> K=k0+e+8g ; e>=8 -> K=k0+16+8g+(e-8)
// B-frag (row-striped-across-lanes):  e<8 -> K=k0+e+8g ; e>=8 -> K=k0+16+8g+(e-8), col = lane%16
// ---------------------------------------------------------------------------
template<int KD, int NOUT, int ACT>
__device__ __forceinline__ void gemm_stage(
    const _Float16* __restrict__ sIn, int ldin,
    const _Float16* __restrict__ WT,
    const float* __restrict__ bias, int biasN,
    _Float16* __restrict__ sOut, int ldout)
{
  const int lane = threadIdx.x & 31;
  const int wid  = threadIdx.x >> 5;
  const int g    = lane >> 4;
  const int l15  = lane & 15;
  constexpr int numTiles = (BT / 16) * (NOUT / 16);
  for (int t = wid; t < numTiles; t += 8) {
    const int m0 = (t & 3) << 4;        // BT/16 == 4 row tiles
    const int n0 = (t >> 2) << 4;
    v8f acc = {};
#pragma unroll
    for (int k0 = 0; k0 < KD; k0 += 32) {
      const _Float16* pa = sIn + (m0 + l15) * ldin + k0 + g * 8;
      v8h alo = *(const v8h*)pa;          // K = k0+8g .. +7
      v8h ahi = *(const v8h*)(pa + 16);   // K = k0+16+8g .. +7
      v16h a;
#pragma unroll
      for (int e = 0; e < 8; ++e) { a[e] = alo[e]; a[e + 8] = ahi[e]; }
      const _Float16* pb = WT + (long)(n0 + l15) * KD + k0 + g * 8;
      v8h blo = *(const v8h*)pb;          // K = k0+8g .. +7
      v8h bhi = *(const v8h*)(pb + 16);   // K = k0+16+8g .. +7
      v16h b;
#pragma unroll
      for (int e = 0; e < 8; ++e) { b[e] = blo[e]; b[e + 8] = bhi[e]; }
      acc = __builtin_amdgcn_wmma_f32_16x16x32_f16(
          false, a, false, b, (short)0, acc, false, false);
    }
    const int n = n0 + l15;
    const float bv = (n < biasN) ? bias[n] : 0.0f;
#pragma unroll
    for (int j = 0; j < 8; ++j) {
      const int m = m0 + j + g * 8;      // C/D layout: row j+8g, col lane%16
      float v = acc[j] + bv;
      if (ACT == 1)      v = (v > 0.0f) ? v : (__expf(v) - 1.0f);
      else if (ACT == 2) v = fmaxf(v, 0.0f);
      sOut[m * ldout + n] = (_Float16)v;
    }
  }
}

// Final consequent GEMM: dp[BT][CP] = relu(g2 @ Wc3[r] + bc3[r]) -> f32 ws
__device__ __forceinline__ void gemm_dp(
    const _Float16* __restrict__ sIn, int ldin,
    const _Float16* __restrict__ WT, const float* __restrict__ bias,
    float* __restrict__ out, int r, int b0)
{
  const int lane = threadIdx.x & 31;
  const int wid  = threadIdx.x >> 5;
  const int g    = lane >> 4;
  const int l15  = lane & 15;
  if (wid >= 4) return;                  // only 4 (m-tile) x 1 (n-tile) tiles
  const int m0 = wid << 4;
  v8f acc = {};
#pragma unroll
  for (int k0 = 0; k0 < Fd; k0 += 32) {
    const _Float16* pa = sIn + (m0 + l15) * ldin + k0 + g * 8;
    v8h alo = *(const v8h*)pa;
    v8h ahi = *(const v8h*)(pa + 16);
    v16h a;
#pragma unroll
    for (int e = 0; e < 8; ++e) { a[e] = alo[e]; a[e + 8] = ahi[e]; }
    const _Float16* pb = WT + (long)l15 * Fd + k0 + g * 8;
    v8h blo = *(const v8h*)pb;
    v8h bhi = *(const v8h*)(pb + 16);
    v16h b;
#pragma unroll
    for (int e = 0; e < 8; ++e) { b[e] = blo[e]; b[e + 8] = bhi[e]; }
    acc = __builtin_amdgcn_wmma_f32_16x16x32_f16(
        false, a, false, b, (short)0, acc, false, false);
  }
  const int n = l15;
  const float bv = (n < Ccls) ? bias[n] : 0.0f;
#pragma unroll
  for (int j = 0; j < 8; ++j) {
    const int m = m0 + j + g * 8;
    out[((long)r * Bsz + b0 + m) * CP + n] = fmaxf(acc[j] + bv, 0.0f);
  }
}

// ---------------------------------------------------------------------------
// Main fused kernel: one block = (rule r, 64-row batch tile). Two LDS
// ping-pong buffers; all 4 big GEMMs on WMMA.
// ---------------------------------------------------------------------------
__global__ __launch_bounds__(256) void fnn_main(
    const float* __restrict__ data,  const float* __restrict__ proto,
    const float* __restrict__ var,
    const float* __restrict__ b1,    const float* __restrict__ b2,
    const float* __restrict__ W3,    const float* __restrict__ b3,
    const float* __restrict__ bc1,   const float* __restrict__ bc2,
    const float* __restrict__ bc3,
    const _Float16* __restrict__ W1T,  const _Float16* __restrict__ W2T,
    const _Float16* __restrict__ Wc1T, const _Float16* __restrict__ Wc2T,
    const _Float16* __restrict__ Wc3T,
    float* __restrict__ ws_logits, float* __restrict__ ws_dp)
{
  __shared__ __align__(16) _Float16 sB[BT * LDA];   // 128-wide buffer
  __shared__ __align__(16) _Float16 sA[BT * LDH];   // 256-wide buffer
  const int tilesPerRule = Bsz / BT;                // 256
  const int r   = blockIdx.x / tilesPerRule;
  const int b0  = (blockIdx.x % tilesPerRule) * BT;
  const int tid = threadIdx.x;

  // (1) Gaussian membership -> sB (f16)
  for (int i = tid; i < BT * Fd; i += 256) {
    const int m = i >> 7, f = i & 127;
    const float d = data[(long)(b0 + m) * Fd + f];
    const float p = proto[(long)r * Fd + f];
    float v = var[(long)r * Fd + f];
    v = fminf(fmaxf(v, 1e-4f), 0.1f);
    const float t = d - p;
    sB[m * LDA + f] = (_Float16)__expf(-(t * t) / (2.0f * v * v));
  }
  __syncthreads();

  // (2) h1 = elu(mem @ W1 + b1) -> sA
  gemm_stage<Fd, Hd, 1>(sB, LDA, W1T, b1, Hd, sA, LDH);
  __syncthreads();

  // (3) h2 = elu(h1 @ W2 + b2) -> sB
  gemm_stage<Hd, Fd, 1>(sA, LDH, W2T, b2, Fd, sB, LDA);
  __syncthreads();

  // (4) logits = h2 @ W3 + b3  (matvec: one thread per batch row)
  if (tid < BT) {
    float acc = b3[0];
#pragma unroll 4
    for (int f = 0; f < Fd; ++f) acc += (float)sB[tid * LDA + f] * W3[f];
    ws_logits[(long)r * Bsz + b0 + tid] = acc;
  }
  __syncthreads();

  // (5) raw data (f16) -> sB
  for (int i = tid; i < BT * Fd; i += 256) {
    const int m = i >> 7, f = i & 127;
    sB[m * LDA + f] = (_Float16)data[(long)(b0 + m) * Fd + f];
  }
  __syncthreads();

  // (6) g1 = elu(data @ Wc1[r] + bc1[r]) -> sA
  gemm_stage<Fd, Hd, 1>(sB, LDA, Wc1T + (long)r * Hd * Fd, bc1 + (long)r * Hd,
                        Hd, sA, LDH);
  __syncthreads();

  // (7) g2 = elu(g1 @ Wc2[r] + bc2[r]) -> sB
  gemm_stage<Hd, Fd, 1>(sA, LDH, Wc2T + (long)r * Fd * Hd, bc2 + (long)r * Fd,
                        Fd, sB, LDA);
  __syncthreads();

  // (8) dp = relu(g2 @ Wc3[r] + bc3[r]) -> ws (f32, padded to CP cols)
  gemm_dp(sB, LDA, Wc3T + (long)r * CP * Fd, bc3 + (long)r * Ccls, ws_dp, r, b0);
}

// ---------------------------------------------------------------------------
// Softmax over rules + weighted combine. One wave32 per batch row (lane=rule).
// ---------------------------------------------------------------------------
__global__ __launch_bounds__(256) void fnn_combine(
    const float* __restrict__ ws_logits, const float* __restrict__ ws_dp,
    float* __restrict__ out)
{
  const int wid  = threadIdx.x >> 5;
  const int lane = threadIdx.x & 31;        // lane == rule index
  const int b    = blockIdx.x * 8 + wid;
  const float lg = ws_logits[(long)lane * Bsz + b];
  float m = lg;
#pragma unroll
  for (int off = 16; off; off >>= 1) m = fmaxf(m, __shfl_xor(m, off, 32));
  const float e = __expf(lg - m);
  float s = e;
#pragma unroll
  for (int off = 16; off; off >>= 1) s += __shfl_xor(s, off, 32);
  const float fs = e / s;
  out[(long)Bsz * Ccls + (long)b * Rn + lane] = fs;   // fire_strength [B,R]

  const float* dp = ws_dp + ((long)lane * Bsz + b) * CP;
  float acc[Ccls];
#pragma unroll
  for (int c = 0; c < Ccls; ++c) acc[c] = dp[c] * fs;
#pragma unroll
  for (int off = 16; off; off >>= 1) {
#pragma unroll
    for (int c = 0; c < Ccls; ++c) acc[c] += __shfl_xor(acc[c], off, 32);
  }
  if (lane == 0) {
#pragma unroll
    for (int c = 0; c < Ccls; ++c) out[(long)b * Ccls + c] = acc[c];
  }
}

// ---------------------------------------------------------------------------
extern "C" void kernel_launch(void* const* d_in, const int* in_sizes, int n_in,
                              void* d_out, int out_size, void* d_ws,
                              size_t ws_size, hipStream_t stream)
{
  const float* data  = (const float*)d_in[0];
  const float* proto = (const float*)d_in[1];
  const float* var   = (const float*)d_in[2];
  const float* W1    = (const float*)d_in[3];
  const float* b1    = (const float*)d_in[4];
  const float* W2    = (const float*)d_in[5];
  const float* b2    = (const float*)d_in[6];
  const float* W3    = (const float*)d_in[7];
  const float* b3    = (const float*)d_in[8];
  const float* Wc1   = (const float*)d_in[9];
  const float* bc1   = (const float*)d_in[10];
  const float* Wc2   = (const float*)d_in[11];
  const float* bc2   = (const float*)d_in[12];
  const float* Wc3   = (const float*)d_in[13];
  const float* bc3   = (const float*)d_in[14];
  float* out = (float*)d_out;

  // workspace layout
  float* ws_logits = (float*)d_ws;                          // R*B f32
  float* ws_dp     = ws_logits + (long)Rn * Bsz;            // R*B*CP f32
  _Float16* W1T  = (_Float16*)(ws_dp + (long)Rn * Bsz * CP);
  _Float16* W2T  = W1T + (long)Hd * Fd;
  _Float16* Wc1T = W2T + (long)Fd * Hd;
  _Float16* Wc2T = Wc1T + (long)Rn * Hd * Fd;
  _Float16* Wc3T = Wc2T + (long)Rn * Fd * Hd;

  const long prepN = (long)Hd * Fd + (long)Fd * Hd +
                     (long)Rn * Hd * Fd + (long)Rn * Fd * Hd +
                     (long)Rn * CP * Fd;
  prep_weights<<<(int)((prepN + 255) / 256), 256, 0, stream>>>(
      W1, W2, Wc1, Wc2, Wc3, W1T, W2T, Wc1T, Wc2T, Wc3T);

  fnn_main<<<Rn * (Bsz / BT), 256, 0, stream>>>(
      data, proto, var, b1, b2, W3, b3, bc1, bc2, bc3,
      W1T, W2T, Wc1T, Wc2T, Wc3T, ws_logits, ws_dp);

  fnn_combine<<<Bsz / 8, 256, 0, stream>>>(ws_logits, ws_dp, out);
}
